// DecoherenceEngine_12395275616278
// MI455X (gfx1250) — matrix operand order repaired and checked by reading, not compiled, not run on hardware
//
#include <hip/hip_runtime.h>
#include <math.h>

typedef unsigned short u16;
typedef __attribute__((ext_vector_type(16))) __bf16 v16bf;
typedef __attribute__((ext_vector_type(8)))  __bf16 v8bf;
typedef __attribute__((ext_vector_type(8)))  float  v8f;

#define DEV static __device__ __forceinline__

// ---------------- bf16 split helpers (RNE) ----------------
DEV u16 f2bf(float f) {
  unsigned u = __float_as_uint(f);
  u += 0x7fffu + ((u >> 16) & 1u);
  return (u16)(u >> 16);
}
DEV float bf2f(u16 h) { return __uint_as_float(((unsigned)h) << 16); }
DEV void store_pair(u16* hi, u16* lo, size_t idx, float v) {
  u16 h = f2bf(v);
  hi[idx] = h;
  lo[idx] = f2bf(v - bf2f(h));
}

// ---------------- WMMA fragment loaders (TN layout, K-contiguous rows) ----
// A frag (16x32 bf16): lane<16 -> M=lane, K chunks {0..7, 16..23};
//                      lane>=16 -> M=lane-16, K chunks {8..15, 24..31}.
DEV v16bf load_a(const u16* A, int lda, int m0, int k0) {
  int lane = threadIdx.x & 31;
  int half = lane >> 4;
  const u16* p = A + (size_t)(m0 + (lane & 15)) * lda + k0 + half * 8;
  union { struct { v8bf c0, c1; } s; v16bf v; } u;
  u.s.c0 = *(const v8bf*)p;
  u.s.c1 = *(const v8bf*)(p + 16);
  return u.v;
}
// B frag (32x16 bf16), B stored as W[j, k] row-major (j = output col):
// lane holds col n=lane&15, K chunks {0..7, 16..23} (lane<16) / {8..15, 24..31} (lane>=16).
DEV v16bf load_b(const u16* B, int ldb, int n0, int k0) {
  int lane = threadIdx.x & 31;
  int half = lane >> 4;
  const u16* p = B + (size_t)(n0 + (lane & 15)) * ldb + k0 + half * 8;
  union { struct { v8bf c0, c1; } s; v16bf v; } u;
  u.s.c0 = *(const v8bf*)p;
  u.s.c1 = *(const v8bf*)(p + 16);
  return u.v;
}
DEV v8f wmma_bf16(v16bf a, v16bf b, v8f c) {
  return __builtin_amdgcn_wmma_f32_16x16x32_bf16(false, a, false, b, (short)0, c,
                                                 false, false);
}
// hi*hi + hi*lo + lo*hi  (drop lo*lo; rel err ~2^-18)
DEV v8f mm3(v16bf ah, v16bf al, v16bf bh, v16bf bl, v8f c) {
  c = wmma_bf16(ah, bh, c);
  c = wmma_bf16(ah, bl, c);
  c = wmma_bf16(al, bh, c);
  return c;
}
// prefetch the next K block of this lane's A row (global_prefetch_b8)
DEV void prefetch_a(const u16* A, int lda, int m0, int k0) {
  int lane = threadIdx.x & 31;
  const u16* p = A + (size_t)(m0 + (lane & 15)) * lda + k0;
  __builtin_prefetch((const void*)p, 0, 0);
}

DEV float sigmoidf_(float x) { return 1.0f / (1.0f + expf(-x)); }

// ================= conversion / prep kernels ==================
__global__ void k_cvt(const float* __restrict__ src, u16* hi, u16* lo, int n) {
  int i = blockIdx.x * blockDim.x + threadIdx.x;
  if (i < n) store_pair(hi, lo, i, src[i]);
}

// stacked layer-1 weights: rows 0..127 = a_W1[:,512:1536], 128..255 = g_W1[:,512:1536]
__global__ void k_cvt_w1s(const float* __restrict__ aW1, const float* __restrict__ gW1,
                          u16* hi, u16* lo) {
  int i = blockIdx.x * blockDim.x + threadIdx.x;  // 256*1024
  if (i >= 256 * 1024) return;
  int j = i >> 10, k = i & 1023;
  float v = (j < 128) ? aW1[j * 1536 + 512 + k] : gW1[(j - 128) * 1536 + 512 + k];
  store_pair(hi, lo, i, v);
}

// stacked layer-2 weights: [j, e] = e<128 ? a_W2[j,e] : -g_W2[j,e-128]
__global__ void k_cvt_w2s(const float* __restrict__ aW2, const float* __restrict__ gW2,
                          u16* hi, u16* lo) {
  int i = blockIdx.x * blockDim.x + threadIdx.x;  // 512*256
  if (i >= 512 * 256) return;
  int j = i >> 8, e = i & 255;
  float v = (e < 128) ? aW2[j * 128 + e] : -gW2[j * 128 + (e - 128)];
  store_pair(hi, lo, i, v);
}

// gru_Wih[3072,513] -> first 512 cols as bf16 pair; last col (tension weight) fp32
__global__ void k_cvt_wih(const float* __restrict__ Wih, u16* hi, u16* lo,
                          float* __restrict__ wlast) {
  int i = blockIdx.x * blockDim.x + threadIdx.x;  // 3072*512
  if (i < 3072) wlast[i] = Wih[(size_t)i * 513 + 512];
  if (i >= 3072 * 512) return;
  int j = i >> 9, k = i & 511;
  store_pair(hi, lo, i, Wih[(size_t)j * 513 + k]);
}

// effective MLP biases (fold constant x-contribution) + b2 difference
__global__ void k_small_pre(const float* __restrict__ x,
                            const float* __restrict__ aW1, const float* __restrict__ ab1,
                            const float* __restrict__ gW1, const float* __restrict__ gb1,
                            const float* __restrict__ ab2, const float* __restrict__ gb2,
                            float* be, float* b2d) {
  int t = threadIdx.x;  // 512 threads, 1 block
  if (t < 256) {
    const float* wrow = (t < 128) ? (aW1 + (size_t)t * 1536) : (gW1 + (size_t)(t - 128) * 1536);
    float s = (t < 128) ? ab1[t] : gb1[t - 128];
    for (int k = 0; k < 512; ++k) s += x[k] * wrow[k];
    be[t] = s;
  }
  b2d[t] = ab2[t] - gb2[t];
}

// decohere: softmax(rho_diag + noise*g*0.1) | abs(rho_off*(1-g)) -> feat bf16 pair
__global__ void k_prep(const float* __restrict__ rho_d, const float* __restrict__ noise_d,
                       const float* __restrict__ rho_o, const float* __restrict__ gamma,
                       u16* fh, u16* fl) {
  int n = blockIdx.x, t = threadIdx.x;  // 8192 blocks x 256
  float g = sigmoidf_(gamma[0]);
  __shared__ float red[256];
  float v[4];
  float mx = -3.4e38f;
#pragma unroll
  for (int i = 0; i < 4; ++i) {
    int idx = t + i * 256;
    v[i] = rho_d[(size_t)n * 1024 + idx] + noise_d[(size_t)n * 1024 + idx] * g * 0.1f;
    mx = fmaxf(mx, v[i]);
  }
  red[t] = mx; __syncthreads();
  for (int s = 128; s > 0; s >>= 1) { if (t < s) red[t] = fmaxf(red[t], red[t + s]); __syncthreads(); }
  mx = red[0]; __syncthreads();
  float se = 0.0f;
#pragma unroll
  for (int i = 0; i < 4; ++i) { v[i] = expf(v[i] - mx); se += v[i]; }
  red[t] = se; __syncthreads();
  for (int s = 128; s > 0; s >>= 1) { if (t < s) red[t] += red[t + s]; __syncthreads(); }
  float inv = 1.0f / red[0];
  float og = 1.0f - g;
#pragma unroll
  for (int i = 0; i < 4; ++i) {
    int idx = t + i * 256;
    size_t base = (size_t)n * 2048;
    store_pair(fh, fl, base + idx, v[i] * inv);
    store_pair(fh, fl, base + 1024 + idx, fabsf(rho_o[(size_t)n * 1024 + idx] * og));
  }
}

// ================= GEMM kernels (one 32x64 tile per wave) ==================
// GEMM1: inj = feat @ proj_W^T ; h = hiddens + 0.1*(inj + proj_b)
__global__ void __launch_bounds__(256, 1)
k_gemm1(const u16* __restrict__ Ah, const u16* __restrict__ Al,
        const u16* __restrict__ Bh, const u16* __restrict__ Bl,
        const float* __restrict__ pb, const float* __restrict__ hiddens,
        float* __restrict__ h, u16* __restrict__ hhi, u16* __restrict__ hlo) {
  const int K = 2048;
  int wid = blockIdx.x * (blockDim.x >> 5) + (threadIdx.x >> 5);
  int nw = wid & 15, mw = wid >> 4;
  int m0 = mw * 32, n0 = nw * 64;
  v8f acc[2][4] = {};
  for (int k0 = 0; k0 < K; k0 += 32) {
    if (k0 + 32 < K) {
      prefetch_a(Ah, K, m0, k0 + 32);
      prefetch_a(Al, K, m0, k0 + 32);
    }
    v16bf ah0 = load_a(Ah, K, m0, k0),      ah1 = load_a(Ah, K, m0 + 16, k0);
    v16bf al0 = load_a(Al, K, m0, k0),      al1 = load_a(Al, K, m0 + 16, k0);
#pragma unroll
    for (int j = 0; j < 4; ++j) {
      v16bf bh = load_b(Bh, K, n0 + j * 16, k0);
      v16bf bl = load_b(Bl, K, n0 + j * 16, k0);
      acc[0][j] = mm3(ah0, al0, bh, bl, acc[0][j]);
      acc[1][j] = mm3(ah1, al1, bh, bl, acc[1][j]);
    }
  }
  int lane = threadIdx.x & 31, half = lane >> 4, cl = lane & 15;
#pragma unroll
  for (int ms = 0; ms < 2; ++ms)
#pragma unroll
    for (int j = 0; j < 4; ++j)
#pragma unroll
      for (int r = 0; r < 8; ++r) {
        int row = m0 + ms * 16 + r + 8 * half;
        int col = n0 + j * 16 + cl;
        size_t o = (size_t)row * 1024 + col;
        float hv = hiddens[o] + 0.1f * (acc[ms][j][r] + pb[col]);
        h[o] = hv;
        store_pair(hhi, hlo, o, hv);
      }
}

// GEMM2: hid = relu(h @ W1s^T + be)   [N=256, K=1024]
__global__ void __launch_bounds__(256, 1)
k_gemm2(const u16* __restrict__ Ah, const u16* __restrict__ Al,
        const u16* __restrict__ Bh, const u16* __restrict__ Bl,
        const float* __restrict__ be, u16* __restrict__ hid_h,
        u16* __restrict__ hid_l) {
  const int K = 1024;
  int wid = blockIdx.x * (blockDim.x >> 5) + (threadIdx.x >> 5);
  int nw = wid & 3, mw = wid >> 2;
  int m0 = mw * 32, n0 = nw * 64;
  v8f acc[2][4] = {};
  for (int k0 = 0; k0 < K; k0 += 32) {
    v16bf ah0 = load_a(Ah, K, m0, k0),      ah1 = load_a(Ah, K, m0 + 16, k0);
    v16bf al0 = load_a(Al, K, m0, k0),      al1 = load_a(Al, K, m0 + 16, k0);
#pragma unroll
    for (int j = 0; j < 4; ++j) {
      v16bf bh = load_b(Bh, K, n0 + j * 16, k0);
      v16bf bl = load_b(Bl, K, n0 + j * 16, k0);
      acc[0][j] = mm3(ah0, al0, bh, bl, acc[0][j]);
      acc[1][j] = mm3(ah1, al1, bh, bl, acc[1][j]);
    }
  }
  int lane = threadIdx.x & 31, half = lane >> 4, cl = lane & 15;
#pragma unroll
  for (int ms = 0; ms < 2; ++ms)
#pragma unroll
    for (int j = 0; j < 4; ++j)
#pragma unroll
      for (int r = 0; r < 8; ++r) {
        int row = m0 + ms * 16 + r + 8 * half;
        int col = n0 + j * 16 + cl;
        float v = fmaxf(acc[ms][j][r] + be[col], 0.0f);
        store_pair(hid_h, hid_l, (size_t)row * 256 + col, v);
      }
}

// GEMM3: out = hid @ W2s^T + b2d ; tension = mean(out^2) via atomics  [N=512, K=256]
__global__ void __launch_bounds__(256, 1)
k_gemm3(const u16* __restrict__ Ah, const u16* __restrict__ Al,
        const u16* __restrict__ Bh, const u16* __restrict__ Bl,
        const float* __restrict__ b2d, float* __restrict__ outp,
        u16* __restrict__ ohi, u16* __restrict__ olo,
        float* __restrict__ tension) {
  const int K = 256;
  int wid = blockIdx.x * (blockDim.x >> 5) + (threadIdx.x >> 5);
  int nw = wid & 7, mw = wid >> 3;
  int m0 = mw * 32, n0 = nw * 64;
  v8f acc[2][4] = {};
  for (int k0 = 0; k0 < K; k0 += 32) {
    v16bf ah0 = load_a(Ah, K, m0, k0),      ah1 = load_a(Ah, K, m0 + 16, k0);
    v16bf al0 = load_a(Al, K, m0, k0),      al1 = load_a(Al, K, m0 + 16, k0);
#pragma unroll
    for (int j = 0; j < 4; ++j) {
      v16bf bh = load_b(Bh, K, n0 + j * 16, k0);
      v16bf bl = load_b(Bl, K, n0 + j * 16, k0);
      acc[0][j] = mm3(ah0, al0, bh, bl, acc[0][j]);
      acc[1][j] = mm3(ah1, al1, bh, bl, acc[1][j]);
    }
  }
  int lane = threadIdx.x & 31, half = lane >> 4, cl = lane & 15;
#pragma unroll
  for (int ms = 0; ms < 2; ++ms)
#pragma unroll
    for (int r = 0; r < 8; ++r) {
      int row = m0 + ms * 16 + r + 8 * half;
      float s = 0.0f;
#pragma unroll
      for (int j = 0; j < 4; ++j) {
        int col = n0 + j * 16 + cl;
        float v = acc[ms][j][r] + b2d[col];
        size_t o = (size_t)row * 512 + col;
        outp[o] = v;
        store_pair(ohi, olo, o, v);
        s += v * v;
      }
      s += __shfl_xor(s, 1);
      s += __shfl_xor(s, 2);
      s += __shfl_xor(s, 4);
      s += __shfl_xor(s, 8);
      if (cl == 0) atomicAdd(&tension[row], s * (1.0f / 512.0f));
    }
}

// Fused GRU: gi (K=512, out) + gh (K=1024, h) for gate rows {j, j+1024, j+2048},
// full gate math in epilogue; writes pre-sync new_h directly into d_out region.
__global__ void __launch_bounds__(256, 1)
k_gru(const u16* __restrict__ OutH, const u16* __restrict__ OutL,
      const u16* __restrict__ Hh, const u16* __restrict__ Hl,
      const u16* __restrict__ WihH, const u16* __restrict__ WihL,
      const u16* __restrict__ WhhH, const u16* __restrict__ WhhL,
      const float* __restrict__ bih, const float* __restrict__ bhh,
      const float* __restrict__ wlast, const float* __restrict__ tension,
      const float* __restrict__ h, float* __restrict__ newh) {
  int wid = blockIdx.x * (blockDim.x >> 5) + (threadIdx.x >> 5);
  int jw = wid & 63, mw = wid >> 6;
  int m0 = mw * 32, j0 = jw * 16;
  v8f ar[2] = {}, az[2] = {}, ain[2] = {}, ahn[2] = {};
  // pass 1: gi = out @ Wih[:, :512]^T
  for (int k0 = 0; k0 < 512; k0 += 32) {
    if (k0 + 32 < 512) {
      prefetch_a(OutH, 512, m0, k0 + 32);
      prefetch_a(OutL, 512, m0, k0 + 32);
    }
    v16bf ah0 = load_a(OutH, 512, m0, k0),      ah1 = load_a(OutH, 512, m0 + 16, k0);
    v16bf al0 = load_a(OutL, 512, m0, k0),      al1 = load_a(OutL, 512, m0 + 16, k0);
    v16bf bh, bl;
    bh = load_b(WihH, 512, j0, k0);        bl = load_b(WihL, 512, j0, k0);
    ar[0] = mm3(ah0, al0, bh, bl, ar[0]);  ar[1] = mm3(ah1, al1, bh, bl, ar[1]);
    bh = load_b(WihH, 512, j0 + 1024, k0); bl = load_b(WihL, 512, j0 + 1024, k0);
    az[0] = mm3(ah0, al0, bh, bl, az[0]);  az[1] = mm3(ah1, al1, bh, bl, az[1]);
    bh = load_b(WihH, 512, j0 + 2048, k0); bl = load_b(WihL, 512, j0 + 2048, k0);
    ain[0] = mm3(ah0, al0, bh, bl, ain[0]); ain[1] = mm3(ah1, al1, bh, bl, ain[1]);
  }
  // pass 2: gh = h @ Whh^T  (r,z sum with gi; n kept separate)
  for (int k0 = 0; k0 < 1024; k0 += 32) {
    if (k0 + 32 < 1024) {
      prefetch_a(Hh, 1024, m0, k0 + 32);
      prefetch_a(Hl, 1024, m0, k0 + 32);
    }
    v16bf ah0 = load_a(Hh, 1024, m0, k0),      ah1 = load_a(Hh, 1024, m0 + 16, k0);
    v16bf al0 = load_a(Hl, 1024, m0, k0),      al1 = load_a(Hl, 1024, m0 + 16, k0);
    v16bf bh, bl;
    bh = load_b(WhhH, 1024, j0, k0);        bl = load_b(WhhL, 1024, j0, k0);
    ar[0] = mm3(ah0, al0, bh, bl, ar[0]);   ar[1] = mm3(ah1, al1, bh, bl, ar[1]);
    bh = load_b(WhhH, 1024, j0 + 1024, k0); bl = load_b(WhhL, 1024, j0 + 1024, k0);
    az[0] = mm3(ah0, al0, bh, bl, az[0]);   az[1] = mm3(ah1, al1, bh, bl, az[1]);
    bh = load_b(WhhH, 1024, j0 + 2048, k0); bl = load_b(WhhL, 1024, j0 + 2048, k0);
    ahn[0] = mm3(ah0, al0, bh, bl, ahn[0]); ahn[1] = mm3(ah1, al1, bh, bl, ahn[1]);
  }
  int lane = threadIdx.x & 31, half = lane >> 4, cl = lane & 15;
#pragma unroll
  for (int ms = 0; ms < 2; ++ms)
#pragma unroll
    for (int r = 0; r < 8; ++r) {
      int row = m0 + ms * 16 + r + 8 * half;
      int col = j0 + cl;
      float t = tension[row];
      float pr = ar[ms][r]  + bih[col]        + bhh[col]        + t * wlast[col];
      float pz = az[ms][r]  + bih[col + 1024] + bhh[col + 1024] + t * wlast[col + 1024];
      float pin = ain[ms][r] + bih[col + 2048]                  + t * wlast[col + 2048];
      float phn = ahn[ms][r] + bhh[col + 2048];
      float rg = sigmoidf_(pr);
      float zg = sigmoidf_(pz);
      float ng = tanhf(pin + rg * phn);
      size_t o = (size_t)row * 1024 + col;
      newh[o] = (1.0f - zg) * ng + zg * h[o];
    }
}

// ================= faction sync/debate + mixture ==================
__global__ void k_fmean(const float* __restrict__ newh, float* fmean) {
  int i = blockIdx.x * blockDim.x + threadIdx.x;  // 8192 = 8 factions x 1024 cols
  int f = i >> 10, c = i & 1023;
  float s = 0.0f;
  for (int r = 0; r < 1024; ++r) s += newh[((size_t)f * 1024 + r) * 1024 + c];
  fmean[i] = s * (1.0f / 1024.0f);
}
__global__ void k_glob(const float* __restrict__ fmean, float* glob) {
  int c = blockIdx.x * blockDim.x + threadIdx.x;  // 1024
  float s = 0.0f;
  for (int f = 0; f < 8; ++f) s += fmean[f * 1024 + c];
  glob[c] = s * 0.125f;
}
__global__ void k_combine(float* newh, const float* __restrict__ fmean,
                          const float* __restrict__ glob, const int* __restrict__ step) {
  size_t i = (size_t)blockIdx.x * blockDim.x + threadIdx.x;
  if (i >= (size_t)8192 * 1024) return;
  int n = (int)(i >> 10), c = (int)(i & 1023);
  float v = newh[i];
  v = 0.85f * v + 0.15f * fmean[(n >> 10) * 1024 + c];
  if (step[0] > 5 && (n & 1023) < 256) v = 0.85f * v + 0.15f * glob[c];
  newh[i] = v;
}
__global__ void k_softw(const float* __restrict__ tension, float* w) {
  int t = threadIdx.x;  // 1 block x 256
  __shared__ float red[256];
  float mx = -3.4e38f;
  for (int i = t; i < 8192; i += 256) mx = fmaxf(mx, tension[i]);
  red[t] = mx; __syncthreads();
  for (int s = 128; s > 0; s >>= 1) { if (t < s) red[t] = fmaxf(red[t], red[t + s]); __syncthreads(); }
  mx = red[0]; __syncthreads();
  float se = 0.0f;
  for (int i = t; i < 8192; i += 256) se += expf(tension[i] - mx);
  red[t] = se; __syncthreads();
  for (int s = 128; s > 0; s >>= 1) { if (t < s) red[t] += red[t + s]; __syncthreads(); }
  float inv = 1.0f / red[0];
  for (int i = t; i < 8192; i += 256) w[i] = expf(tension[i] - mx) * inv;
}
__global__ void k_combined(const float* __restrict__ w, const float* __restrict__ outp,
                           float* comb) {
  int c = blockIdx.x * blockDim.x + threadIdx.x;  // 512
  if (c >= 512) return;
  float s = 0.0f;
  for (int n = 0; n < 8192; ++n) s += w[n] * outp[(size_t)n * 512 + c];
  comb[c] = s;
}
__global__ void k_pred(const float* __restrict__ comb, const float* __restrict__ headW,
                       const float* __restrict__ headb, float* pred) {
  int i = blockIdx.x * blockDim.x + threadIdx.x;  // 512
  if (i >= 512) return;
  float s = headb[i];
  for (int c = 0; c < 512; ++c) s += comb[c] * headW[(size_t)i * 512 + c];
  pred[i] = s;
}

// ======================= launcher =======================
extern "C" void kernel_launch(void* const* d_in, const int* in_sizes, int n_in,
                              void* d_out, int out_size, void* d_ws, size_t ws_size,
                              hipStream_t stream) {
  const float* x        = (const float*)d_in[0];
  const float* rho_diag = (const float*)d_in[1];
  const float* rho_off  = (const float*)d_in[2];
  const float* hiddens  = (const float*)d_in[3];
  const float* gamma    = (const float*)d_in[4];
  const float* proj_W   = (const float*)d_in[5];
  const float* proj_b   = (const float*)d_in[6];
  const float* a_W1     = (const float*)d_in[7];
  const float* a_b1     = (const float*)d_in[8];
  const float* a_W2     = (const float*)d_in[9];
  const float* a_b2     = (const float*)d_in[10];
  const float* g_W1     = (const float*)d_in[11];
  const float* g_b1     = (const float*)d_in[12];
  const float* g_W2     = (const float*)d_in[13];
  const float* g_b2     = (const float*)d_in[14];
  const float* gru_Wih  = (const float*)d_in[15];
  const float* gru_bih  = (const float*)d_in[16];
  const float* gru_Whh  = (const float*)d_in[17];
  const float* gru_bhh  = (const float*)d_in[18];
  const float* head_W   = (const float*)d_in[19];
  const float* head_b   = (const float*)d_in[20];
  const float* noise_d  = (const float*)d_in[21];
  // d_in[22] = noise_off (unused by the reference math)
  const int*   step     = (const int*)d_in[23];

  const size_t N = 8192;
  char* ws = (char*)d_ws;
  size_t off = 0;
  auto alloc = [&](size_t bytes) {
    char* p = ws + off;
    off = (off + bytes + 255) & ~(size_t)255;
    return p;
  };
  u16*   feat_hi = (u16*)alloc(N * 2048 * 2);
  u16*   feat_lo = (u16*)alloc(N * 2048 * 2);
  float* h_f     = (float*)alloc(N * 1024 * 4);
  u16*   h_hi    = (u16*)alloc(N * 1024 * 2);
  u16*   h_lo    = (u16*)alloc(N * 1024 * 2);
  u16*   pw_hi   = (u16*)alloc(1024 * 2048 * 2);
  u16*   pw_lo   = (u16*)alloc(1024 * 2048 * 2);
  u16*   w1_hi   = (u16*)alloc(256 * 1024 * 2);
  u16*   w1_lo   = (u16*)alloc(256 * 1024 * 2);
  u16*   w2_hi   = (u16*)alloc(512 * 256 * 2);
  u16*   w2_lo   = (u16*)alloc(512 * 256 * 2);
  u16*   wih_hi  = (u16*)alloc(3072 * 512 * 2);
  u16*   wih_lo  = (u16*)alloc(3072 * 512 * 2);
  float* wih_last= (float*)alloc(3072 * 4);
  u16*   whh_hi  = (u16*)alloc(3072 * 1024 * 2);
  u16*   whh_lo  = (u16*)alloc(3072 * 1024 * 2);
  u16*   hid_hi  = (u16*)alloc(N * 256 * 2);
  u16*   hid_lo  = (u16*)alloc(N * 256 * 2);
  float* out_f   = (float*)alloc(N * 512 * 4);
  u16*   out_hi  = (u16*)alloc(N * 512 * 2);
  u16*   out_lo  = (u16*)alloc(N * 512 * 2);
  float* tension = (float*)alloc(N * 4);
  float* be      = (float*)alloc(256 * 4);
  float* b2d     = (float*)alloc(512 * 4);
  float* fmean   = (float*)alloc(8 * 1024 * 4);
  float* glob    = (float*)alloc(1024 * 4);
  float* wprob   = (float*)alloc(N * 4);
  float* comb    = (float*)alloc(512 * 4);

  float* pred = (float*)d_out;          // out[0:512]
  float* newh = pred + 512;             // out[512 : 512 + 8192*1024]

  // --- weight conversion / precompute (independent, serialized on stream) ---
  k_cvt<<<(1024 * 2048 + 255) / 256, 256, 0, stream>>>(proj_W, pw_hi, pw_lo, 1024 * 2048);
  k_cvt<<<(3072 * 1024 + 255) / 256, 256, 0, stream>>>(gru_Whh, whh_hi, whh_lo, 3072 * 1024);
  k_cvt_w1s<<<(256 * 1024 + 255) / 256, 256, 0, stream>>>(a_W1, g_W1, w1_hi, w1_lo);
  k_cvt_w2s<<<(512 * 256 + 255) / 256, 256, 0, stream>>>(a_W2, g_W2, w2_hi, w2_lo);
  k_cvt_wih<<<(3072 * 512 + 255) / 256, 256, 0, stream>>>(gru_Wih, wih_hi, wih_lo, wih_last);
  k_small_pre<<<1, 512, 0, stream>>>(x, a_W1, a_b1, g_W1, g_b1, a_b2, g_b2, be, b2d);
  k_prep<<<8192, 256, 0, stream>>>(rho_diag, noise_d, rho_off, gamma, feat_hi, feat_lo);
  hipMemsetAsync(tension, 0, N * 4, stream);

  // --- GEMM chain ---
  k_gemm1<<<512, 256, 0, stream>>>(feat_hi, feat_lo, pw_hi, pw_lo, proj_b, hiddens,
                                   h_f, h_hi, h_lo);
  k_gemm2<<<128, 256, 0, stream>>>(h_hi, h_lo, w1_hi, w1_lo, be, hid_hi, hid_lo);
  k_gemm3<<<256, 256, 0, stream>>>(hid_hi, hid_lo, w2_hi, w2_lo, b2d, out_f,
                                   out_hi, out_lo, tension);
  k_gru<<<2048, 256, 0, stream>>>(out_hi, out_lo, h_hi, h_lo, wih_hi, wih_lo,
                                  whh_hi, whh_lo, gru_bih, gru_bhh, wih_last,
                                  tension, h_f, newh);

  // --- faction sync/debate (in place on d_out new_h region) ---
  k_fmean<<<32, 256, 0, stream>>>(newh, fmean);
  k_glob<<<4, 256, 0, stream>>>(fmean, glob);
  k_combine<<<(int)((N * 1024 + 255) / 256), 256, 0, stream>>>(newh, fmean, glob, step);

  // --- tension-weighted mixture + head ---
  k_softw<<<1, 256, 0, stream>>>(tension, wprob);
  k_combined<<<2, 256, 0, stream>>>(wprob, out_f, comb);
  k_pred<<<2, 256, 0, stream>>>(comb, head_W, head_b, pred);
}
